// RNNBlock_20005957665327
// MI455X (gfx1250) — matrix operand order repaired and checked by reading, not compiled, not run on hardware
//
#include <hip/hip_runtime.h>

// ---------------------------------------------------------------------------
// LRU block (diagonal complex linear RNN) for MI455X / gfx1250, wave32 + WMMA
// + TDM (tensor_load_to_lds) double-buffered tile staging.
//   B=16, L=4096, D_OUT=1024, H=512.  Both GEMMs: 65536 x 1024 x 1024 bf16.
// ---------------------------------------------------------------------------

typedef __attribute__((ext_vector_type(16))) __bf16 v16bf;
typedef __attribute__((ext_vector_type(8)))  __bf16 v8bf;
typedef __attribute__((ext_vector_type(8)))  float  v8f;
typedef __attribute__((ext_vector_type(4)))  unsigned int v4u;
typedef __attribute__((ext_vector_type(8)))  int v8i;
typedef __attribute__((ext_vector_type(4)))  int v4i;

#define BATCH   16
#define SEQL    4096
#define HDIM    512
#define DOUT    1024
#define MTOT    (BATCH * SEQL)   // 65536
#define KDIM    1024             // GEMM K (both GEMMs)
#define NDIM    1024             // GEMM N (both GEMMs)
#define TCHUNK  128
#define NCHUNK  (SEQL / TCHUNK)  // 32
#define KSTEPS  (KDIM / 32)      // 32

// ---- workspace layout (bytes) --------------------------------------------
static const size_t OFF_W1     = 0;                                   // 2 MiB bf16 (N,K)
static const size_t OFF_W2     = OFF_W1 + (size_t)NDIM * KDIM * 2;    // 2 MiB bf16 (N,K)
static const size_t OFF_LAM    = OFF_W2 + (size_t)NDIM * KDIM * 2;    // 4 KiB float2
static const size_t OFF_LAMPOW = OFF_LAM + (size_t)HDIM * 8;          // 512 KiB float2 (lam^(k+1))
static const size_t OFF_CARRY  = OFF_LAMPOW + (size_t)TCHUNK * HDIM * 8; // 2 MiB float2
static const size_t OFF_PREFIX = OFF_CARRY + (size_t)BATCH * NCHUNK * HDIM * 8;
static const size_t OFF_BUH    = OFF_PREFIX + (size_t)BATCH * NCHUNK * HDIM * 8; // 128 MiB bf16
// total ~136.5 MiB

__device__ __forceinline__ float2 cmul(float2 a, float2 b) {
    return make_float2(a.x * b.x - a.y * b.y, a.x * b.y + a.y * b.x);
}

// ---------------------------------------------------------------------------
// TDM: DMA a 2D bf16 tile (tile_d1 rows x tile_d0 elems, row stride = stride_d0
// elements) from global memory into LDS at lds_addr, packed row-major.
// D# layout per CDNA5 ISA (2D: groups 2,3 = 0).
// 6-arg builtin variant (clang-23 / therock-10.0 headers).
// ---------------------------------------------------------------------------
__device__ __forceinline__ void tdm_load_tile_bf16(
        unsigned lds_addr, const __bf16* gptr,
        unsigned tensor_d0, unsigned tensor_d1, unsigned stride_d0,
        unsigned tile_d0, unsigned tile_d1) {
    unsigned long long ga = (unsigned long long)(uintptr_t)gptr;
    v4u g0;
    g0[0] = 1u;                                       // count=1 (valid), user mode
    g0[1] = lds_addr;                                 // LDS byte address
    g0[2] = (unsigned)(ga & 0xFFFFFFFFu);             // global_addr[31:0]
    g0[3] = (unsigned)((ga >> 32) & 0x01FFFFFFu)      // global_addr[56:32]
          | (2u << 30);                               // type=2 ("image")
    v8i g1;
    g1[0] = (int)(1u << 16);                          // data_size=1 (2 bytes)
    g1[1] = (int)((tensor_d0 & 0xFFFFu) << 16);       // tensor_dim0[15:0]
    g1[2] = (int)(((tensor_d0 >> 16) & 0xFFFFu)       // tensor_dim0[31:16]
          | ((tensor_d1 & 0xFFFFu) << 16));           // tensor_dim1[15:0]
    g1[3] = (int)(((tensor_d1 >> 16) & 0xFFFFu)       // tensor_dim1[31:16]
          | ((tile_d0 & 0xFFFFu) << 16));             // tile_dim0
    g1[4] = (int)(tile_d1 & 0xFFFFu);                 // tile_dim1 (tile_dim2=0)
    g1[5] = (int)stride_d0;                           // tensor_dim0_stride[31:0]
    g1[6] = 0;                                        // stride[47:32]=0, dim1_stride lo=0
    g1[7] = 0;
    v4i z4 = {0, 0, 0, 0};
    v8i z8 = {0, 0, 0, 0, 0, 0, 0, 0};
    __builtin_amdgcn_tensor_load_to_lds(g0, g1, z4, z4, z8, 0);
}

// ---------------------------------------------------------------------------
// prep: W1[n,k] = bf16( (n<H ? B_re[n,k] : B_im[n-H,k]) * exp(gamma_log[h]) )
// ---------------------------------------------------------------------------
__global__ __launch_bounds__(256) void prep_w1_kernel(
        const float* __restrict__ Bre, const float* __restrict__ Bim,
        const float* __restrict__ gamma_log, __bf16* __restrict__ W1) {
    int t = blockIdx.x * 256 + threadIdx.x;      // 0 .. N*K-1
    int n = t >> 10;
    int k = t & 1023;
    int h = (n < HDIM) ? n : (n - HDIM);
    float g = __expf(gamma_log[h]);
    float v = (n < HDIM) ? Bre[(size_t)h * KDIM + k] : Bim[(size_t)h * KDIM + k];
    W1[t] = (__bf16)(v * g);
}

// W2[o,k] = bf16( k<H ? C_re[o,k] : -C_im[o,k-H] )
__global__ __launch_bounds__(256) void prep_w2_kernel(
        const float* __restrict__ Cre, const float* __restrict__ Cim,
        __bf16* __restrict__ W2) {
    int t = blockIdx.x * 256 + threadIdx.x;
    int o = t >> 10;
    int k = t & 1023;
    float v = (k < HDIM) ? Cre[(size_t)o * HDIM + k]
                         : -Cim[(size_t)o * HDIM + (k - HDIM)];
    W2[t] = (__bf16)v;
}

// lam[h] = exp(-exp(nu) + i*exp(theta));  lampow[k][h] = lam^(k+1), k=0..T-1
__global__ __launch_bounds__(256) void prep_lambda_kernel(
        const float* __restrict__ theta_log, const float* __restrict__ nu_log,
        float2* __restrict__ lam, float2* __restrict__ lampow) {
    int h = blockIdx.x * 256 + threadIdx.x;
    if (h >= HDIM) return;
    float nu = __expf(nu_log[h]);
    float th = __expf(theta_log[h]);
    float r  = __expf(-nu);
    float2 lm = make_float2(r * __cosf(th), r * __sinf(th));
    lam[h] = lm;
    float2 p = make_float2(1.0f, 0.0f);
    for (int k = 0; k < TCHUNK; ++k) {
        p = cmul(p, lm);
        lampow[(size_t)k * HDIM + h] = p;
    }
}

// ---------------------------------------------------------------------------
// Fragment gather helpers (CDNA5 WMMA 16-bit VGPR layouts).
//  A 16x32: lane<16 -> M=lane, K in {0..7,16..23}; lane>=16 -> M=lane-16, +8.
//  B 32x16: lane<16 -> N=lane, K in {0..7,16..23}; lane>=16 -> N=lane-16, +8.
// ---------------------------------------------------------------------------
__device__ __forceinline__ v16bf load_afrag(const __bf16* As, int row, int kA) {
    const __bf16* base = As + row * 32 + kA;
    v8bf lo = *(const v8bf*)base;
    v8bf hi = *(const v8bf*)(base + 16);
    return __builtin_shufflevector(lo, hi, 0,1,2,3,4,5,6,7,8,9,10,11,12,13,14,15);
}
__device__ __forceinline__ v16bf load_bfrag(const __bf16* Bs, int col, int kB) {
    const __bf16* base = Bs + col * 32 + kB;
    v8bf lo = *(const v8bf*)base;
    v8bf hi = *(const v8bf*)(base + 16);
    return __builtin_shufflevector(lo, hi, 0,1,2,3,4,5,6,7,8,9,10,11,12,13,14,15);
}

// ---------------------------------------------------------------------------
// GEMM1: Bu[r,n] = bf16( sum_k x[r,k] * W1[n,k] )
//   A tile: f32 -> bf16 converted while staging (manual path).
//   B tile: bf16 weights DMA'd by the Tensor Data Mover, double buffered.
// block 128x128, 8 waves (4x2), wave tile 32x64, K-step 32.
// ---------------------------------------------------------------------------
__global__ __launch_bounds__(256) void gemm1_kernel(
        const float* __restrict__ X, const __bf16* __restrict__ W,
        __bf16* __restrict__ Out) {
    __shared__ __bf16 As[2][128 * 32];
    __shared__ __bf16 Bs[2][128 * 32];
    const int tid  = threadIdx.x;
    const int lane = tid & 31;
    const int wave = tid >> 5;
    const int wm   = wave >> 1;              // 0..3
    const int wn   = wave & 1;               // 0..1
    const int rowBase = blockIdx.y * 128;
    const int colBase = blockIdx.x * 128;
    const int ldRow = tid >> 1;              // 0..127
    const int ldK0  = (tid & 1) * 16;

    const unsigned ldsB[2] = { (unsigned)(uintptr_t)(&Bs[0][0]),
                               (unsigned)(uintptr_t)(&Bs[1][0]) };
    const __bf16* wTileBase = W + (size_t)colBase * KDIM;

    v8f acc[2][4] = {};

    // ---- prologue: stage tile 0 (A manual, B via TDM) ----
    {
        const float* src = X + (size_t)(rowBase + ldRow) * KDIM + ldK0;
        __bf16* dst = &As[0][ldRow * 32 + ldK0];
        #pragma unroll
        for (int q = 0; q < 4; ++q) {
            float4 f = ((const float4*)src)[q];
            dst[q * 4 + 0] = (__bf16)f.x;
            dst[q * 4 + 1] = (__bf16)f.y;
            dst[q * 4 + 2] = (__bf16)f.z;
            dst[q * 4 + 3] = (__bf16)f.w;
        }
    }
    if (tid == 0)
        tdm_load_tile_bf16(ldsB[0], wTileBase, KDIM, NDIM, KDIM, 32, 128);
    __builtin_amdgcn_s_wait_tensorcnt(0);
    __syncthreads();

    const int lm16 = lane & 15;
    const int kA = (lane >> 4) * 8;
    const int kB = (lane >> 4) * 8;

    for (int it = 0; it < KSTEPS; ++it) {
        const int cur = it & 1;
        const bool has_next = (it + 1) < KSTEPS;

        // prefetch next B tile via TDM (wave 0 issues; EXEC-ignored DMA)
        if (has_next && tid == 0)
            tdm_load_tile_bf16(ldsB[cur ^ 1], wTileBase + (it + 1) * 32,
                               KDIM, NDIM, KDIM, 32, 128);
        // issue next A-tile global loads early
        float4 fa[4];
        if (has_next) {
            const float* src = X + (size_t)(rowBase + ldRow) * KDIM + (it + 1) * 32 + ldK0;
            #pragma unroll
            for (int q = 0; q < 4; ++q) fa[q] = ((const float4*)src)[q];
        }

        // ---- compute on current buffers ----
        v16bf afrag[2];
        #pragma unroll
        for (int i = 0; i < 2; ++i)
            afrag[i] = load_afrag(&As[cur][0], wm * 32 + i * 16 + lm16, kA);
        v16bf bfrag[4];
        #pragma unroll
        for (int j = 0; j < 4; ++j)
            bfrag[j] = load_bfrag(&Bs[cur][0], wn * 64 + j * 16 + lm16, kB);
        #pragma unroll
        for (int i = 0; i < 2; ++i)
            #pragma unroll
            for (int j = 0; j < 4; ++j)
                acc[i][j] = __builtin_amdgcn_wmma_f32_16x16x32_bf16(
                    false, afrag[i], false, bfrag[j], (short)0, acc[i][j], false, false);

        if (has_next) {
            // convert + stage next A tile into the other buffer
            __bf16* dst = &As[cur ^ 1][ldRow * 32 + ldK0];
            #pragma unroll
            for (int q = 0; q < 4; ++q) {
                dst[q * 4 + 0] = (__bf16)fa[q].x;
                dst[q * 4 + 1] = (__bf16)fa[q].y;
                dst[q * 4 + 2] = (__bf16)fa[q].z;
                dst[q * 4 + 3] = (__bf16)fa[q].w;
            }
            __builtin_amdgcn_s_wait_tensorcnt(0);   // no-op for non-issuing waves
            __syncthreads();
        }
    }

    const int mhalf = (lane >> 4) * 8;
    const int nloc  = lane & 15;
    #pragma unroll
    for (int i = 0; i < 2; ++i)
        #pragma unroll
        for (int j = 0; j < 4; ++j) {
            int col = colBase + wn * 64 + j * 16 + nloc;
            #pragma unroll
            for (int v = 0; v < 8; ++v) {
                int row = rowBase + wm * 32 + i * 16 + mhalf + v;
                Out[(size_t)row * NDIM + col] = (__bf16)acc[i][j][v];
            }
        }
}

// ---------------------------------------------------------------------------
// Scan pass A: per (b,chunk,h) local complex scan, in place on BuH (bf16),
// f32 accumulation; writes chunk carry.
// ---------------------------------------------------------------------------
__global__ __launch_bounds__(256) void scan_a_kernel(
        __bf16* __restrict__ BuH, const float2* __restrict__ lam,
        float2* __restrict__ carry) {
    int t = blockIdx.x * 256 + threadIdx.x;      // B*NC*H threads
    int h = t & (HDIM - 1);
    int c = (t >> 9) & (NCHUNK - 1);
    int b = t >> 14;
    float2 lm = lam[h];
    float hre = 0.0f, him = 0.0f;
    int rowBase = b * SEQL + c * TCHUNK;
    for (int k = 0; k < TCHUNK; ++k) {
        size_t base = (size_t)(rowBase + k) * NDIM;
        float bre = (float)BuH[base + h];
        float bim = (float)BuH[base + HDIM + h];
        float nre = lm.x * hre - lm.y * him + bre;
        float nim = lm.x * him + lm.y * hre + bim;
        hre = nre; him = nim;
        BuH[base + h]        = (__bf16)hre;
        BuH[base + HDIM + h] = (__bf16)him;
    }
    carry[((size_t)b * NCHUNK + c) * HDIM + h] = make_float2(hre, him);
}

// Scan pass B: per (b,h) sequential scan over NC chunk carries -> exclusive prefixes
__global__ __launch_bounds__(256) void scan_b_kernel(
        const float2* __restrict__ carry, const float2* __restrict__ lampow,
        float2* __restrict__ prefix) {
    int t = blockIdx.x * 256 + threadIdx.x;      // B*H threads
    int h = t & (HDIM - 1);
    int b = t >> 9;
    float2 lamT = lampow[(size_t)(TCHUNK - 1) * HDIM + h];   // lam^T
    float2 run = make_float2(0.0f, 0.0f);
    for (int c = 0; c < NCHUNK; ++c) {
        size_t idx = ((size_t)b * NCHUNK + c) * HDIM + h;
        prefix[idx] = run;
        float2 cv = carry[idx];
        float2 pr = cmul(lamT, run);
        run = make_float2(pr.x + cv.x, pr.y + cv.y);
    }
}

// Scan pass C: h_t = lam^(k+1) * prefix + local; in place; emit h[:, -1] f32.
__global__ __launch_bounds__(256) void scan_c_kernel(
        __bf16* __restrict__ BuH, const float2* __restrict__ lampow,
        const float2* __restrict__ prefix, float* __restrict__ hlast) {
    int t = blockIdx.x * 256 + threadIdx.x;      // B*NC*H threads
    int h = t & (HDIM - 1);
    int c = (t >> 9) & (NCHUNK - 1);
    int b = t >> 14;
    float2 p = prefix[((size_t)b * NCHUNK + c) * HDIM + h];
    int rowBase = b * SEQL + c * TCHUNK;
    for (int k = 0; k < TCHUNK; ++k) {
        size_t base = (size_t)(rowBase + k) * NDIM;
        float2 lp = lampow[(size_t)k * HDIM + h];   // lam^(k+1)
        float lre = (float)BuH[base + h];
        float lim = (float)BuH[base + HDIM + h];
        float2 inj = cmul(lp, p);
        float hre = lre + inj.x;
        float him = lim + inj.y;
        BuH[base + h]        = (__bf16)hre;
        BuH[base + HDIM + h] = (__bf16)him;
        if (c == NCHUNK - 1 && k == TCHUNK - 1) {
            size_t o = ((size_t)b * HDIM + h) * 2;  // complex64: interleaved re,im
            hlast[o]     = hre;
            hlast[o + 1] = him;
        }
    }
}

// ---------------------------------------------------------------------------
// GEMM2: y[r,o] = sum_k H[r,k] * W2[o,k] + D[o]*x[r,o]
//   Both tiles bf16 in global -> both staged by TDM, double buffered.
// ---------------------------------------------------------------------------
__global__ __launch_bounds__(256) void gemm2_kernel(
        const __bf16* __restrict__ Hb, const __bf16* __restrict__ W,
        const float* __restrict__ Dv, const float* __restrict__ X,
        float* __restrict__ Y) {
    __shared__ __bf16 As[2][128 * 32];
    __shared__ __bf16 Bs[2][128 * 32];
    const int tid  = threadIdx.x;
    const int lane = tid & 31;
    const int wave = tid >> 5;
    const int wm   = wave >> 1;
    const int wn   = wave & 1;
    const int rowBase = blockIdx.y * 128;
    const int colBase = blockIdx.x * 128;

    const unsigned ldsA[2] = { (unsigned)(uintptr_t)(&As[0][0]),
                               (unsigned)(uintptr_t)(&As[1][0]) };
    const unsigned ldsB[2] = { (unsigned)(uintptr_t)(&Bs[0][0]),
                               (unsigned)(uintptr_t)(&Bs[1][0]) };
    const __bf16* aTileBase = Hb + (size_t)rowBase * KDIM;
    const __bf16* wTileBase = W  + (size_t)colBase * KDIM;

    v8f acc[2][4] = {};

    if (tid == 0) {
        tdm_load_tile_bf16(ldsA[0], aTileBase, KDIM, MTOT, KDIM, 32, 128);
        tdm_load_tile_bf16(ldsB[0], wTileBase, KDIM, NDIM, KDIM, 32, 128);
    }
    __builtin_amdgcn_s_wait_tensorcnt(0);
    __syncthreads();

    const int lm16 = lane & 15;
    const int kA = (lane >> 4) * 8;
    const int kB = (lane >> 4) * 8;

    for (int it = 0; it < KSTEPS; ++it) {
        const int cur = it & 1;
        const bool has_next = (it + 1) < KSTEPS;

        if (has_next && tid == 0) {
            tdm_load_tile_bf16(ldsA[cur ^ 1], aTileBase + (it + 1) * 32,
                               KDIM, MTOT, KDIM, 32, 128);
            tdm_load_tile_bf16(ldsB[cur ^ 1], wTileBase + (it + 1) * 32,
                               KDIM, NDIM, KDIM, 32, 128);
        }

        v16bf afrag[2];
        #pragma unroll
        for (int i = 0; i < 2; ++i)
            afrag[i] = load_afrag(&As[cur][0], wm * 32 + i * 16 + lm16, kA);
        v16bf bfrag[4];
        #pragma unroll
        for (int j = 0; j < 4; ++j)
            bfrag[j] = load_bfrag(&Bs[cur][0], wn * 64 + j * 16 + lm16, kB);
        #pragma unroll
        for (int i = 0; i < 2; ++i)
            #pragma unroll
            for (int j = 0; j < 4; ++j)
                acc[i][j] = __builtin_amdgcn_wmma_f32_16x16x32_bf16(
                    false, afrag[i], false, bfrag[j], (short)0, acc[i][j], false, false);

        if (has_next) {
            __builtin_amdgcn_s_wait_tensorcnt(0);
            __syncthreads();
        }
    }

    const int mhalf = (lane >> 4) * 8;
    const int nloc  = lane & 15;
    #pragma unroll
    for (int i = 0; i < 2; ++i)
        #pragma unroll
        for (int j = 0; j < 4; ++j) {
            int col = colBase + wn * 64 + j * 16 + nloc;
            float dcol = Dv[col];
            #pragma unroll
            for (int v = 0; v < 8; ++v) {
                int row = rowBase + wm * 32 + i * 16 + mhalf + v;
                size_t idx = (size_t)row * NDIM + col;
                Y[idx] = acc[i][j][v] + dcol * X[idx];
            }
        }
}

// ---------------------------------------------------------------------------
extern "C" void kernel_launch(void* const* d_in, const int* in_sizes, int n_in,
                              void* d_out, int out_size, void* d_ws, size_t ws_size,
                              hipStream_t stream) {
    (void)in_sizes; (void)n_in; (void)out_size; (void)ws_size;
    const float* x         = (const float*)d_in[0];
    const float* theta_log = (const float*)d_in[1];
    const float* nu_log    = (const float*)d_in[2];
    const float* gamma_log = (const float*)d_in[3];
    const float* B_re      = (const float*)d_in[4];
    const float* B_im      = (const float*)d_in[5];
    const float* C_re      = (const float*)d_in[6];
    const float* C_im      = (const float*)d_in[7];
    const float* Dv        = (const float*)d_in[8];

    char* ws = (char*)d_ws;
    __bf16* W1      = (__bf16*)(ws + OFF_W1);
    __bf16* W2      = (__bf16*)(ws + OFF_W2);
    float2* lam     = (float2*)(ws + OFF_LAM);
    float2* lampow  = (float2*)(ws + OFF_LAMPOW);
    float2* carry   = (float2*)(ws + OFF_CARRY);
    float2* prefix  = (float2*)(ws + OFF_PREFIX);
    __bf16* BuH     = (__bf16*)(ws + OFF_BUH);

    float* y     = (float*)d_out;
    float* hlast = (float*)d_out + (size_t)MTOT * DOUT;   // complex64 tail

    prep_w1_kernel<<<(NDIM * KDIM) / 256, 256, 0, stream>>>(B_re, B_im, gamma_log, W1);
    prep_w2_kernel<<<(NDIM * KDIM) / 256, 256, 0, stream>>>(C_re, C_im, W2);
    prep_lambda_kernel<<<2, 256, 0, stream>>>(theta_log, nu_log, lam, lampow);

    dim3 gemm_grid(NDIM / 128, MTOT / 128);   // (8, 512)
    gemm1_kernel<<<gemm_grid, 256, 0, stream>>>(x, W1, BuH);

    scan_a_kernel<<<(BATCH * NCHUNK * HDIM) / 256, 256, 0, stream>>>(BuH, lam, carry);
    scan_b_kernel<<<(BATCH * HDIM) / 256, 256, 0, stream>>>(carry, lampow, prefix);
    scan_c_kernel<<<(BATCH * NCHUNK * HDIM) / 256, 256, 0, stream>>>(BuH, lampow, prefix, hlast);

    gemm2_kernel<<<gemm_grid, 256, 0, stream>>>(BuH, W2, Dv, x, y);
}
